// CRF_Loss_18966575579727
// MI455X (gfx1250) — matrix operand order repaired and checked by reading, not compiled, not run on hardware
//
#include <hip/hip_runtime.h>
#include <hip/hip_bf16.h>

typedef __attribute__((ext_vector_type(16))) _Float16 v16h;
typedef __attribute__((ext_vector_type(8)))  _Float16 v8h;
typedef __attribute__((ext_vector_type(8)))  float    v8f;

#define B_     64
#define T_     1024
#define K_     256
#define NT_    258
#define START_ 256
#define END_   257

#define LOG2E_ 1.44269504088896340f
#define LN2_   0.69314718055994531f

// ---------------------------------------------------------------------------
// Kernel 1: CRF forward recursion (log-partition) as exp-space WMMA matmuls.
// grid = 4 blocks (16 batch rows each), block = 512 threads = 16 wave32s.
// Phase B: wave w owns alpha row w -> shfl-butterfly rowmax, p=exp2((a-m)*log2e)
//          written as ONE contiguous ds_store_b128 in WMMA-A layout.
// Phase A: wave w owns N-tile w -> 8x v_wmma_f32_16x16x32_f16 against resident
//          exp(T) B-fragments, then log via raw v_log_f32 + branchless blend.
// 2 barriers per step.
// ---------------------------------------------------------------------------
__global__ __launch_bounds__(512) void crf_forward(const float* __restrict__ em,
                                                   const float* __restrict__ trans,
                                                   const int*   __restrict__ tags,
                                                   float*       __restrict__ la_out)
{
    __shared__ float s_la[16 * 256];                 // 16 KB  log_alpha (f32)
    __shared__ __align__(32) _Float16 s_p[8 * 512];  // 8 KB   A-fragments (f16)
    __shared__ __align__(16) float s_rowmax[16];
    __shared__ __align__(16) float s_mask[16];

    const int tid  = threadIdx.x;
    const int lane = tid & 31;       // wave32 lane
    const int w    = tid >> 5;       // wave id 0..15 (= N-tile, = alpha row)
    const int hi   = lane >> 4;      // lane half (A/B layout)
    const int ml   = lane & 15;
    const int bb   = blockIdx.x * 16;  // batch base row

    // ---- build persistent B fragments: exp(T[k][n]) in f16 (once) ----
    // B 32x16 f16 fragment in this code's contiguous k order: lane l holds column n = l&15; K = (l>>4)*16 + h.
    v16h bfrag[8];
#pragma unroll
    for (int kt = 0; kt < 8; ++kt) {
#pragma unroll
        for (int h = 0; h < 16; ++h) {
            const int k = kt * 32 + hi * 16 + h;
            const int n0 = w * 16 + ml;
            bfrag[kt][h] = (_Float16)__expf(trans[k * NT_ + n0]);
        }
    }

    // phase-B addressing: wave w = row w, lane covers cols [lane*8, lane*8+8)
    const int col0 = lane * 8;
    // A-layout target for this lane's 8 contiguous k values (all in one b128):
    //   kt = lane>>2, lane' = ((lane>>1)&1)*16 + w, h_base = (lane&1) ? 8 : 0
    const int pOff = (lane >> 2) * 512 + ((((lane >> 1) & 1) << 4) + w) * 16 +
                     ((lane & 1) ? 8 : 0);

    // ---- init log_alpha(t=0) = em[:,0,:] + T[START,:] ----
#pragma unroll
    for (int j = 0; j < 8; ++j) {
        const int c = col0 + j;
        s_la[w * 256 + c] = em[(bb + w) * T_ * K_ + c] + trans[START_ * NT_ + c];
    }
    __syncthreads();

    // per-wave register copy of log_alpha for this wave's 16 N-columns
    const int n = w * 16 + ml;
    float la_reg[8];
#pragma unroll
    for (int r = 0; r < 8; ++r)
        la_reg[r] = s_la[(r + 8 * hi) * 256 + n];

    for (int t = 1; t < T_; ++t) {
        // ===== phase B: rowmax (shfl butterfly) + p into A layout =====
        float la8[8];
        float lm = -3.0e38f;
#pragma unroll
        for (int j = 0; j < 8; ++j) {
            la8[j] = s_la[w * 256 + col0 + j];
            lm = fmaxf(lm, la8[j]);
        }
#pragma unroll
        for (int off = 16; off > 0; off >>= 1)
            lm = fmaxf(lm, __shfl_xor(lm, off, 32));
        if (lane == 0) {
            s_rowmax[w] = lm;
            const int tg = tags[(bb + w) * T_ + t];
            s_mask[w] = (tg != -1) ? 1.0f : 0.0f;
        }
        {
            v8h ph;
#pragma unroll
            for (int j = 0; j < 8; ++j)
                ph[j] = (_Float16)__builtin_amdgcn_exp2f((la8[j] - lm) * LOG2E_);
            *reinterpret_cast<v8h*>(&s_p[pOff]) = ph;
        }
        __syncthreads();

        // ===== phase A: per-wave 16x16x256 slab via 8 WMMAs =====
        v16h afrag[8];
#pragma unroll
        for (int kt = 0; kt < 8; ++kt)
            afrag[kt] = *reinterpret_cast<const v16h*>(&s_p[kt * 512 + lane * 16]);

        v8f c = {};
#pragma unroll
        for (int kt = 0; kt < 8; ++kt)
            c = __builtin_amdgcn_wmma_f32_16x16x32_f16(
                    false, afrag[kt], false, bfrag[kt],
                    (short)0, c, false, false);

        // vectorized rowmax/mask reads for this lane's 8 C/D rows (8*hi..8*hi+7)
        const float4 rm0 = *reinterpret_cast<const float4*>(&s_rowmax[8 * hi]);
        const float4 rm1 = *reinterpret_cast<const float4*>(&s_rowmax[8 * hi + 4]);
        const float4 mk0 = *reinterpret_cast<const float4*>(&s_mask[8 * hi]);
        const float4 mk1 = *reinterpret_cast<const float4*>(&s_mask[8 * hi + 4]);
        const float rm8[8] = {rm0.x, rm0.y, rm0.z, rm0.w, rm1.x, rm1.y, rm1.z, rm1.w};
        const float mk8[8] = {mk0.x, mk0.y, mk0.z, mk0.w, mk1.x, mk1.y, mk1.z, mk1.w};

#pragma unroll
        for (int r = 0; r < 8; ++r) {
            const int m = r + 8 * hi;                 // C/D row = r + 8*(lane>>4)
            const float e = em[((bb + m) * T_ + t) * K_ + n];
            // c >= 256*min(p)*min(expT) > 0 and never subnormal: raw v_log_f32 ok
            const float nv = __builtin_amdgcn_logf(c[r]) * LN2_ + rm8[r] + e;
            la_reg[r] = fmaf(mk8[r], nv - la_reg[r], la_reg[r]);  // branchless blend
            s_la[m * 256 + n] = la_reg[r];
        }
        if (t + 1 < T_) {
#pragma unroll
            for (int r = 0; r < 8; ++r)
                __builtin_prefetch(&em[((bb + r + 8 * hi) * T_ + t + 1) * K_ + n], 0, 1);
        }
        __syncthreads();
    }

    // ---- final log_alpha -> workspace for kernel 2 ----
#pragma unroll
    for (int j = 0; j < 8; ++j) {
        const int c = col0 + j;
        la_out[(bb + w) * K_ + c] = s_la[w * 256 + c];
    }
}

// ---------------------------------------------------------------------------
// Kernel 2: gold-path score (parallel over t) + log Z from final alpha.
// grid = 64 blocks (one per batch row), block = 256 threads.
// ---------------------------------------------------------------------------
__global__ __launch_bounds__(256) void crf_score(const float* __restrict__ em,
                                                 const float* __restrict__ trans,
                                                 const int*   __restrict__ tags,
                                                 const float* __restrict__ la_fin,
                                                 float*       __restrict__ out)
{
    __shared__ float red[256];
    __shared__ int   redi[256];
    const int b = blockIdx.x;
    const int tid = threadIdx.x;

    // path terms + mask count (each thread covers 4 timesteps)
    float psum = 0.0f;
    int mcount = 0;
#pragma unroll
    for (int kk = 0; kk < 4; ++kk) {
        const int t = tid * 4 + kk;
        const int tg = tags[b * T_ + t];
        const int msk = (tg != -1);
        mcount += msk;
        if (t >= 1) {
            int prev = tags[b * T_ + t - 1]; prev = prev < 0 ? 0 : prev;
            int cur = tg < 0 ? 0 : tg;
            const float term = em[(b * T_ + t) * K_ + cur] + trans[prev * NT_ + cur];
            if (msk) psum += term;
        }
    }
    red[tid] = psum;
    redi[tid] = mcount;
    __syncthreads();
    for (int s = 128; s > 0; s >>= 1) {
        if (tid < s) { red[tid] += red[tid + s]; redi[tid] += redi[tid + s]; }
        __syncthreads();
    }
    const float path = red[0];
    const int last_idx = redi[0] - 1;
    __syncthreads();

    // log Z = logsumexp_j(alpha_T[j] + T[j, END])
    const float v = la_fin[b * K_ + tid] + trans[tid * NT_ + END_];
    red[tid] = v;
    __syncthreads();
    for (int s = 128; s > 0; s >>= 1) {
        if (tid < s) red[tid] = fmaxf(red[tid], red[tid + s]);
        __syncthreads();
    }
    const float mx = red[0];
    __syncthreads();
    red[tid] = __expf(v - mx);
    __syncthreads();
    for (int s = 128; s > 0; s >>= 1) {
        if (tid < s) red[tid] += red[tid + s];
        __syncthreads();
    }
    if (tid == 0) {
        const float logz = mx + __logf(red[0]);
        const int t0 = tags[b * T_];
        const int tc0 = t0 < 0 ? 0 : t0;
        const float m0 = (t0 != -1) ? 1.0f : 0.0f;
        const float llh0 = trans[START_ * NT_ + tc0] + em[b * T_ * K_ + tc0] * m0;
        int lt = tags[b * T_ + last_idx]; lt = lt < 0 ? 0 : lt;
        out[b] = path + llh0 + trans[lt * NT_ + END_] - logz;
    }
}

extern "C" void kernel_launch(void* const* d_in, const int* in_sizes, int n_in,
                              void* d_out, int out_size, void* d_ws, size_t ws_size,
                              hipStream_t stream)
{
    (void)in_sizes; (void)n_in; (void)out_size; (void)ws_size;
    const float* em    = (const float*)d_in[0];   // (64,1024,256) f32
    const float* trans = (const float*)d_in[1];   // (258,258) f32
    const int*   tags  = (const int*)d_in[2];     // (64,1024) int
    float* la_ws = (float*)d_ws;                  // 64*256 f32 final alpha
    float* out   = (float*)d_out;                 // (64,) f32

    crf_forward<<<dim3(B_ / 16), dim3(512), 0, stream>>>(em, trans, tags, la_ws);
    crf_score<<<dim3(B_), dim3(256), 0, stream>>>(em, trans, tags, la_ws, out);
}
